// CausalSelfAttention_27376121545421
// MI455X (gfx1250) — matrix operand order repaired and checked by reading, not compiled, not run on hardware
//
#include <hip/hip_runtime.h>

typedef __attribute__((ext_vector_type(16))) _Float16 v16h;
typedef __attribute__((ext_vector_type(8)))  _Float16 v8h;
typedef __attribute__((ext_vector_type(8)))  float    v8f;

// Problem: (B, S, 3, H, D) = (2, 4096, 3, 16, 64), fp32 in/out, causal attention.
constexpr int Bc = 2;
constexpr int Sc = 4096;
constexpr int Hc = 16;
constexpr int Dc = 64;
constexpr int QT = 64;            // query rows per block (16 per wave, 4 waves)
constexpr int WAVES = 4;
constexpr float SCALE = 0.125f;   // 1/sqrt(64), folded into Q

// LDS pitches (halves). 144B / 80B rows: 16B-aligned, bank-rotating.
constexpr int PK = 72;            // sK row pitch
constexpr int PV = 40;            // sVt row pitch
constexpr int PP = 40;            // sP row pitch

__device__ inline v8h cvt8(const float4 a, const float4 b) {
    v8h r;
    r[0] = (_Float16)a.x; r[1] = (_Float16)a.y; r[2] = (_Float16)a.z; r[3] = (_Float16)a.w;
    r[4] = (_Float16)b.x; r[5] = (_Float16)b.y; r[6] = (_Float16)b.z; r[7] = (_Float16)b.w;
    return r;
}

__global__ __launch_bounds__(128)
void fa_fwd_kernel(const float* __restrict__ qkv, float* __restrict__ out) {
    __shared__ __align__(16) _Float16 sK [2][32][PK];
    __shared__ __align__(16) _Float16 sVt[2][64][PV];
    __shared__ __align__(16) _Float16 sP [WAVES][16][PP];

    const int tid  = threadIdx.x;
    const int lane = tid & 31;
    // Force the wave id into an SGPR: all per-wave regime branches below must be
    // SCALAR branches (s_cbranch), never EXEC-masked — WMMA requires EXEC all-ones.
    const int wave = __builtin_amdgcn_readfirstlane(tid >> 5);

    const int nQT = Sc / QT;                 // 64
    const int bid = blockIdx.x;
    const int qt  = bid % nQT;
    const int h   = (bid / nQT) % Hc;
    const int b   = bid / (nQT * Hc);

    const long strideS = 3L * Hc * Dc;       // floats between consecutive seq rows of qkv
    const float* qB = qkv + (long)b * Sc * strideS + 0L * Hc * Dc + (long)h * Dc;
    const float* kB = qkv + (long)b * Sc * strideS + 1L * Hc * Dc + (long)h * Dc;
    const float* vB = qkv + (long)b * Sc * strideS + 2L * Hc * Dc + (long)h * Dc;

    const int row16 = lane & 15;             // 0..15
    const int hi    = lane >> 4;             // half-wave group
    const int kloA  = hi * 8;                // A-layout K offset (interleaved)
    const int kloB  = hi * 8;                // B-layout K offset (interleaved)

    const int qRow0 = qt * QT + wave * 16;   // first query row of this wave (SGPR)
    const int lastW = qRow0 + 15;            // last query row of this wave (SGPR)

    // Cooperative staging: thread -> (key row r, 16-wide dim chunk c0) of the 32x64 tile.
    const int stR  = tid >> 2;               // 0..31
    const int stC0 = (tid & 3) * 16;         // 0,16,32,48

    auto stage = [&](int jt, int buf) {
        const int j0 = jt * 32;
        const float4* kp = (const float4*)(kB + (long)(j0 + stR) * strideS + stC0);
        const float4* vp = (const float4*)(vB + (long)(j0 + stR) * strideS + stC0);
        const float4 k0 = kp[0], k1 = kp[1], k2 = kp[2], k3 = kp[3];
        const float4 w0 = vp[0], w1 = vp[1], w2 = vp[2], w3 = vp[3];
        *(v8h*)&sK[buf][stR][stC0]     = cvt8(k0, k1);
        *(v8h*)&sK[buf][stR][stC0 + 8] = cvt8(k2, k3);
        const v8h vlo = cvt8(w0, w1), vhi = cvt8(w2, w3);
        #pragma unroll
        for (int j = 0; j < 8; ++j) {
            sVt[buf][stC0 + j][stR]     = vlo[j];
            sVt[buf][stC0 + 8 + j][stR] = vhi[j];
        }
    };

    // ---- Load Q as two 16x32 A-fragments (dims 0..31 / 32..63), scale folded in.
    v16h aQ[2];
    {
        const float* qr = qB + (long)(qRow0 + row16) * strideS;
        #pragma unroll
        for (int f = 0; f < 2; ++f) {
            const float* p = qr + f * 32 + kloA;
            v16h a;
            #pragma unroll
            for (int i = 0; i < 8; ++i) a[i]     = (_Float16)(p[i]      * SCALE);
            #pragma unroll
            for (int i = 0; i < 8; ++i) a[8 + i] = (_Float16)(p[16 + i] * SCALE);
            aQ[f] = a;
        }
    }

    // ---- Online-softmax state. Index v corresponds to row qRow0 + v + 8*hi.
    v8f acc[4];
    #pragma unroll
    for (int nd = 0; nd < 4; ++nd)
        #pragma unroll
        for (int v = 0; v < 8; ++v) acc[nd][v] = 0.0f;

    float mrow[8], lrow[8];
    #pragma unroll
    for (int v = 0; v < 8; ++v) { mrow[v] = -1e30f; lrow[v] = 0.0f; }

    // Uniform trip count over the block; per-wave regimes use SCALAR branches.
    const int nKV = ((qt * QT + QT - 1) >> 5) + 1;

    stage(0, 0);
    __syncthreads();

    for (int jt = 0; jt < nKV; ++jt) {
        const int buf = jt & 1;
        const int j0  = jt * 32;

        // Prefetch next tile into the other buffer (overlaps WMMA + softmax below).
        if (jt + 1 < nKV) stage(jt + 1, buf ^ 1);

        // SCALAR branch (j0, lastW both SGPR): tiles entirely above the diagonal
        // contribute nothing for this wave — skip all compute, still hit the barrier.
        if (j0 <= lastW) {
            // ---- K B-fragments from LDS: two 16B loads per fragment (k 8h..8h+7, 16+8h..16+8h+7).
            v16h bK[2][2];
            #pragma unroll
            for (int T = 0; T < 2; ++T)
                #pragma unroll
                for (int f = 0; f < 2; ++f) {
                    const _Float16* p = &sK[buf][row16 + 16 * T][f * 32 + kloB];
                    const v8h lo = *(const v8h*)p;
                    const v8h hh = *(const v8h*)(p + 16);
                    v16h x;
                    #pragma unroll
                    for (int i = 0; i < 8; ++i) { x[i] = lo[i]; x[8 + i] = hh[i]; }
                    bK[T][f] = x;
                }

            // ---- Scores: two 16x16 f32 tiles, each = two K=32 WMMA steps.
            v8f s0, s1;
            #pragma unroll
            for (int v = 0; v < 8; ++v) { s0[v] = 0.0f; s1[v] = 0.0f; }
            s0 = __builtin_amdgcn_wmma_f32_16x16x32_f16(false, aQ[0], false, bK[0][0], (short)0, s0, false, false);
            s0 = __builtin_amdgcn_wmma_f32_16x16x32_f16(false, aQ[1], false, bK[0][1], (short)0, s0, false, false);
            s1 = __builtin_amdgcn_wmma_f32_16x16x32_f16(false, aQ[0], false, bK[1][0], (short)0, s1, false, false);
            s1 = __builtin_amdgcn_wmma_f32_16x16x32_f16(false, aQ[1], false, bK[1][1], (short)0, s1, false, false);

            // ---- Causal mask only for the (single) diagonal-intersecting tile (scalar cond).
            if (j0 + 31 > qRow0) {
                #pragma unroll
                for (int v = 0; v < 8; ++v) {
                    const int srow = qRow0 + v + 8 * hi;
                    s0[v] = (j0 + row16      <= srow) ? s0[v] : -1e9f;
                    s1[v] = (j0 + 16 + row16 <= srow) ? s1[v] : -1e9f;
                }
            }

            // ---- Per-row max reduction (16-lane butterflies per half-wave group).
            float pmax[8];
            #pragma unroll
            for (int v = 0; v < 8; ++v) pmax[v] = fmaxf(s0[v], s1[v]);
            #pragma unroll
            for (int msk = 8; msk >= 1; msk >>= 1)
                #pragma unroll
                for (int v = 0; v < 8; ++v)
                    pmax[v] = fmaxf(pmax[v], __shfl_xor(pmax[v], msk, 32));

            // ---- Online softmax update.
            float rsum[8];
            #pragma unroll
            for (int v = 0; v < 8; ++v) {
                const float mnew  = fmaxf(mrow[v], pmax[v]);
                const float alpha = __expf(mrow[v] - mnew);
                mrow[v] = mnew;
                const float p0 = __expf(s0[v] - mnew);
                const float p1 = __expf(s1[v] - mnew);
                s0[v] = p0; s1[v] = p1;
                rsum[v] = p0 + p1;
                lrow[v] *= alpha;
                #pragma unroll
                for (int nd = 0; nd < 4; ++nd) acc[nd][v] *= alpha;
            }
            #pragma unroll
            for (int msk = 8; msk >= 1; msk >>= 1)
                #pragma unroll
                for (int v = 0; v < 8; ++v)
                    rsum[v] += __shfl_xor(rsum[v], msk, 32);
            #pragma unroll
            for (int v = 0; v < 8; ++v) lrow[v] += rsum[v];

            // ---- Re-stripe P (C layout) -> A layout via per-wave LDS (DS in-order per wave).
            #pragma unroll
            for (int v = 0; v < 8; ++v) {
                sP[wave][v + 8 * hi][row16]      = (_Float16)s0[v];
                sP[wave][v + 8 * hi][16 + row16] = (_Float16)s1[v];
            }
            v16h aP;
            {
                const _Float16* pr = &sP[wave][row16][kloA];
                const v8h lo = *(const v8h*)pr;          // ds_load_b128
                const v8h hh = *(const v8h*)(pr + 16);   // ds_load_b128
                #pragma unroll
                for (int i = 0; i < 8; ++i) { aP[i] = lo[i]; aP[8 + i] = hh[i]; }
            }

            // ---- P(16x32) x V(32x64): B-fragments from transposed LDS V.
            #pragma unroll
            for (int nd = 0; nd < 4; ++nd) {
                const _Float16* p = &sVt[buf][row16 + 16 * nd][kloB];
                const v8h lo = *(const v8h*)p;
                const v8h hh = *(const v8h*)(p + 16);
                v16h bV;
                #pragma unroll
                for (int i = 0; i < 8; ++i) { bV[i] = lo[i]; bV[8 + i] = hh[i]; }
                acc[nd] = __builtin_amdgcn_wmma_f32_16x16x32_f16(false, aP, false, bV, (short)0, acc[nd], false, false);
            }
        }

        __syncthreads();   // staged buffer ready; all reads of current buffer done
    }

    // ---- Normalize and write O. Row v+8*hi, column (lane&15)+16*nd (coalesced per VGPR).
    #pragma unroll
    for (int v = 0; v < 8; ++v) {
        const int srow = qRow0 + v + 8 * hi;
        const float inv = 1.0f / lrow[v];
        float* orow = out + (((long)b * Sc + srow) * Hc + h) * Dc;
        #pragma unroll
        for (int nd = 0; nd < 4; ++nd)
            orow[row16 + 16 * nd] = acc[nd][v] * inv;
    }
}

extern "C" void kernel_launch(void* const* d_in, const int* in_sizes, int n_in,
                              void* d_out, int out_size, void* d_ws, size_t ws_size,
                              hipStream_t stream) {
    (void)in_sizes; (void)n_in; (void)d_ws; (void)ws_size; (void)out_size;
    const float* qkv = (const float*)d_in[0];
    float* out = (float*)d_out;
    const int blocks = Bc * Hc * (Sc / QT);  // 2048
    fa_fwd_kernel<<<blocks, 128, 0, stream>>>(qkv, out);
}
